// ResPACFeaturePyramidBlock_6227702579675
// MI455X (gfx1250) — matrix operand order repaired and checked by reading, not compiled, not run on hardware
//
#include <hip/hip_runtime.h>
#include <stdint.h>

// ---------------------------------------------------------------------------
// ResPAC block for MI455X (gfx1250, wave32, WMMA).
// All GEMMs run on v_wmma_f32_16x16x32_bf16 (f32 accumulate).
// K-dimension is reordered tap-major ([m][tap][c] weights) so each 32-wide
// K-slab has a uniform tap: the LDS-fill becomes base + immediate-offset
// strided loads with no per-element division. Activations carry zero halos,
// so implicit-im2col fetches need no bounds checks. The global fill for the
// next K-slab is software-pipelined behind the current slab's WMMAs.
// PAC stage is a GEMM: out[o,n] = sum_{t,c} W[o,(t,c)] *
//   ( y_up[c, n shifted by tap t] * kern[t,n] ), M=128 K=3200 N=18432.
// ---------------------------------------------------------------------------

typedef __bf16 bf16_t;
typedef bf16_t v16bf __attribute__((ext_vector_type(16)));
typedef float  v8f   __attribute__((ext_vector_type(8)));

union Frag16 { v16bf v; uint4 q[2]; };
union Fill8  { bf16_t b[8]; unsigned short h[8]; uint4 q; };

__device__ __forceinline__ unsigned short f32_to_bf16(float f) {
  union { float f; unsigned u; } v; v.f = f;
  unsigned r = v.u + 0x7FFFu + ((v.u >> 16) & 1u);   // round-to-nearest-even
  return (unsigned short)(r >> 16);
}

// ------------------------- problem constants -------------------------------
#define BB   2
#define CIN  256
#define CO   128
#define HH   48
#define WW   48
#define HWL  (HH*WW)           // 2304
#define NL   (BB*HWL)          // 4608
#define H2   96
#define W2   96
#define HW2  (H2*W2)           // 9216
#define N2   (BB*HW2)          // 18432
#define K1   (CIN*9)           // 2304
#define K2   (CO*9)            // 1152
#define KR   CIN               // 256
#define KF   CO                // 128
#define KP   (CO*25)           // 3200

// padded planes
#define PW1   (WW+2)           // 50
#define PHW1  (PW1*(HH+2))     // 2500
#define PW2P  (W2+4)           // 100
#define PHW2P (PW2P*(H2+4))    // 10000

constexpr int MODE_DIRECT = 0;   // 1x1: k == channel
constexpr int MODE_CONV3  = 1;   // implicit 3x3 over bf16 [C][B][PH][PW], pad 1
constexpr int MODE_PAC    = 2;   // z = y_up(f32 padded, 5x5) * kern(f32)

// ------------------------- pack / permute ----------------------------------
__global__ void f32_to_bf16_k(const float* __restrict__ src,
                              unsigned short* __restrict__ dst, int n) {
  int i = blockIdx.x * blockDim.x + threadIdx.x;
  if (i < n) dst[i] = f32_to_bf16(src[i]);
}

// weights [m][c][taps] f32 -> [m][taps][c] bf16 (tap-major K order)
__global__ void repack_w_k(const float* __restrict__ src,
                           unsigned short* __restrict__ dst,
                           int M, int C, int TAPS) {
  int i = blockIdx.x * blockDim.x + threadIdx.x;
  int total = M * C * TAPS;
  if (i >= total) return;
  int m = i / (C * TAPS);
  int rem = i - m * (C * TAPS);
  int t = rem / C, c = rem - t * C;
  dst[i] = f32_to_bf16(src[(size_t)m * C * TAPS + (size_t)c * TAPS + t]);
}

__global__ void zero_u16_k(unsigned short* __restrict__ p, int n) {
  int i = blockIdx.x * blockDim.x + threadIdx.x;
  if (i < n) p[i] = 0;
}
__global__ void zero_f32_k(float* __restrict__ p, int n) {
  int i = blockIdx.x * blockDim.x + threadIdx.x;
  if (i < n) p[i] = 0.f;
}

// NCHW f32 -> [C][B][H+2p][W+2p] bf16 with zero halo
__global__ void permute_pad_k(const float* __restrict__ src,
                              unsigned short* __restrict__ dst,
                              int B, int C, int H, int W, int pad) {
  const int PW = W + 2 * pad, PH = H + 2 * pad, PHW = PW * PH;
  const int total = B * C * PHW;
  int i = blockIdx.x * blockDim.x + threadIdx.x;
  if (i >= total) return;
  int c = i / (B * PHW);
  int rem = i - c * (B * PHW);
  int b = rem / PHW;
  int pp = rem - b * PHW;
  int hp = pp / PW, wp = pp - hp * PW;
  int h = hp - pad, w = wp - pad;
  unsigned short v = 0;
  if (h >= 0 && h < H && w >= 0 && w < W)
    v = f32_to_bf16(src[((size_t)b * C + c) * (H * W) + h * W + w]);
  dst[i] = v;
}

// ------------------------- WMMA GEMM ---------------------------------------
// Block: 256 threads = 8 waves. Wave w -> M rows [16w,16w+16). Block -> 64 N
// columns as four 16-wide WMMA subtiles. K stepped by 32 through a double-
// buffered LDS tile [n(64)][k(32)] (row stride 40 halves -> aligned b128).
// Requires Cv a multiple of 32 so a K-slab never straddles a tap boundary.
template<int MODE, int Nv, int Kv, int Cv, int Hdv, int Wdv, int PADv, int SPADv>
__global__ void __launch_bounds__(256)
gemm_wmma_k(const unsigned short* __restrict__ Wp,   // bf16 weights [128][Kv] tap-major
            const unsigned short* __restrict__ act,  // bf16 padded [C][B][PH][PW]
            const float* __restrict__ actF,          // f32 padded y_up [B][C][PH][PW]
            const float* __restrict__ kern,          // f32 [B][25][HW]
            const float* __restrict__ bias,          // f32 [128]
            float* __restrict__ out) {               // f32 [B][128][store-plane]
  constexpr int HWd  = Hdv * Wdv;
  constexpr int PW   = Wdv + 2 * PADv;
  constexpr int PHW  = PW * (Hdv + 2 * PADv);
  constexpr int SPW  = Wdv + 2 * SPADv;
  constexpr int SPHW = SPW * (Hdv + 2 * SPADv);
  constexpr int LDSS = 40;
  constexpr int BN = 64;
  __shared__ __align__(16) unsigned short lt[2][BN * LDSS];

  const int n0    = blockIdx.x * BN;
  const int lane  = threadIdx.x & 31;
  const int wv    = threadIdx.x >> 5;
  const int mbase = wv * 16;

  v8f acc[4] = {{}, {}, {}, {}};

  // A-fragment (16-bit A 16x32): lane L holds row (L&15),
  // K range (L&16)?16..31:0..15 (contiguous, matches B).
  const int mrow  = mbase + (lane & 15);
  const int aoff  = (lane & 16) ? 16 : 0;
  // B-fragment: lane L holds col (L&15), K range (L&16)?16..31:0..15.
  const int koffB = (lane & 16) ? 16 : 0;

  // cooperative fill: thread -> 8 consecutive k (same tap, consecutive c)
  const int eid = threadIdx.x * 8;       // 0..2040
  const int nn  = eid >> 5;              // 0..63
  const int fk  = eid & 31;              // 0,8,16,24
  const int n   = n0 + nn;
  const int b   = n / HWd;               // compile-time divisor
  const int p   = n - b * HWd;
  const int hh  = p / Wdv;
  const int ww  = p - hh * Wdv;

  // loop-invariant fetch bases (halo padding => no bounds checks)
  const unsigned short* abase =
      act + (size_t)b * PHW + (size_t)hh * PW + ww +
      (MODE == MODE_DIRECT ? PADv * (PW + 1) : 0);
  const float* ybase = actF + ((size_t)b * Cv) * PHW + (size_t)hh * PW + ww;
  const float* kbase = kern + (size_t)b * 25 * HWd + p;

  auto fetch = [&](int k0) -> Fill8 {
    Fill8 fv;
    const int t  = k0 / Cv;              // uniform tap (Cv power of two)
    const int c0 = (k0 & (Cv - 1)) + fk; // first channel of this thread's 8
    if (MODE == MODE_PAC) {
      const float kv = kbase[(size_t)t * HWd];            // one load per slab
      const int th = t / 5, tw = t - th * 5;              // scalar
      const float* yp = ybase + th * PW + tw + (size_t)c0 * PHW;
#pragma unroll
      for (int j = 0; j < 8; ++j)                         // imm-offset loads
        fv.b[j] = (bf16_t)(yp[(size_t)j * PHW] * kv);
    } else {
      const int toff = (MODE == MODE_CONV3) ? ((t / 3) * PW + (t - (t / 3) * 3)) : 0;
      const unsigned short* ap = abase + toff + (size_t)c0 * (BB * PHW);
#pragma unroll
      for (int j = 0; j < 8; ++j)                         // imm-offset loads
        fv.h[j] = ap[(size_t)j * (BB * PHW)];
    }
    return fv;
  };

  Fill8 fv = fetch(0);
  int buf = 0;
  for (int k0 = 0; k0 < Kv; k0 += 32, buf ^= 1) {
    *(uint4*)&lt[buf][nn * LDSS + fk] = fv.q;
    __syncthreads();                     // single barrier (double-buffered)

    // software pipeline: issue next slab's scattered loads behind the WMMAs
    if (k0 + 32 < Kv) fv = fetch(k0 + 32);

    Frag16 a;
    const unsigned short* arow = Wp + (size_t)mrow * Kv + k0;
    a.q[0] = *(const uint4*)(arow + aoff);
    a.q[1] = *(const uint4*)(arow + 8 + aoff);
    __builtin_prefetch(arow + 32, 0, 3); // next K-slab -> global_prefetch

    Frag16 bfr[4];
#pragma unroll
    for (int s = 0; s < 4; ++s) {        // distinct regs: one ds clause + wait
      const uint4* bp =
          (const uint4*)&lt[buf][(s * 16 + (lane & 15)) * LDSS + koffB];
      bfr[s].q[0] = bp[0]; bfr[s].q[1] = bp[1];
    }
#pragma unroll
    for (int s = 0; s < 4; ++s)          // back-to-back WMMAs
      acc[s] = __builtin_amdgcn_wmma_f32_16x16x32_bf16(
          false, a.v, false, bfr[s].v, (short)0, acc[s], false, false);
  }

  // D layout: VGPR v, lanes 0-15 -> M=v, N=lane; lanes 16-31 -> M=v+8, N=lane-16
  const int mo = mbase + ((lane & 16) ? 8 : 0);
#pragma unroll
  for (int s = 0; s < 4; ++s) {
    const int ns = n0 + s * 16 + (lane & 15);
    const int bs = ns / HWd;
    const int ps = ns - bs * HWd;
    const int h2 = ps / Wdv, w2 = ps - h2 * Wdv;
    float* op = out + ((size_t)bs * CO) * SPHW +
                (size_t)(h2 + SPADv) * SPW + (w2 + SPADv);
#pragma unroll
    for (int v = 0; v < 8; ++v) {
      const int mm = mo + v;
      op[(size_t)mm * SPHW] = acc[s][v] + bias[mm];
    }
  }
}

// ------------------------- instance norm + PReLU (+res) --------------------
__global__ void __launch_bounds__(256)
instnorm_k(const float* __restrict__ x, const float* __restrict__ res,
           const float* __restrict__ gamma, const float* __restrict__ beta,
           const float* __restrict__ alpha,
           float* __restrict__ outF,            // f32 [b][c][HW] (unpadded)
           unsigned short* __restrict__ outB,   // bf16 [c][B][PH][PW] interior
           int B, int C, int H, int W, int pad) {
  __shared__ float sh1[256];
  __shared__ float sh2[256];
  const int HW = H * W;
  const int PW = W + 2 * pad, PHW = PW * (H + 2 * pad);
  const int bc = blockIdx.x;
  const int b = bc / C, c = bc - b * C;
  const float* xp = x + (size_t)bc * HW;

  float s = 0.f, ss = 0.f;
  for (int i = threadIdx.x; i < HW; i += blockDim.x) {
    float v = xp[i]; s += v; ss += v * v;
  }
  sh1[threadIdx.x] = s; sh2[threadIdx.x] = ss;
  __syncthreads();
  for (int st = 128; st > 0; st >>= 1) {
    if ((int)threadIdx.x < st) {
      sh1[threadIdx.x] += sh1[threadIdx.x + st];
      sh2[threadIdx.x] += sh2[threadIdx.x + st];
    }
    __syncthreads();
  }
  const float mean = sh1[0] / (float)HW;
  const float var  = sh2[0] / (float)HW - mean * mean;   // biased (jnp.var)
  const float inv  = rsqrtf(var + 1e-5f) * gamma[c];
  const float bt   = beta[c];
  const float a    = alpha[0];
  const float* rp  = res ? res + (size_t)bc * HW : nullptr;
  unsigned short* ob = outB ? outB + ((size_t)c * B + b) * PHW : nullptr;

  for (int i = threadIdx.x; i < HW; i += blockDim.x) {
    float v = (xp[i] - mean) * inv + bt;
    v = (v >= 0.f) ? v : a * v;                // PReLU
    if (rp)   v += rp[i];                      // + residual 1x1 path
    if (outF) outF[(size_t)bc * HW + i] = v;
    if (ob) {
      int hh = i / W, wn = i - hh * W;
      ob[(hh + pad) * PW + (wn + pad)] = f32_to_bf16(v);
    }
  }
}

// ------------------------- bicubic 2x upsample (a=-0.75) -------------------
// scale=2, align_corners=False -> only two phases: t=0.75 (even), t=0.25 (odd).
// Output written into a radius-2 zero halo layout [B][C][100][100] for PAC.
__global__ void upsample_bicubic_k(const float* __restrict__ y,
                                   float* __restrict__ yu) {
  const int total = BB * CO * PHW2P;
  int idx = blockIdx.x * blockDim.x + threadIdx.x;
  if (idx >= total) return;
  int wp = idx % PW2P; int r = idx / PW2P;
  int hp = r % (H2 + 4); r /= (H2 + 4);       // r = b*CO + c
  int oh = hp - 2, ow = wp - 2;
  if (oh < 0 || oh >= H2 || ow < 0 || ow >= W2) { yu[idx] = 0.f; return; }
  const float* src = y + (size_t)r * HWL;

  const float we[4] = {-0.03515625f, 0.26171875f, 0.87890625f, -0.10546875f}; // t=0.75
  const float wo[4] = {-0.10546875f, 0.87890625f, 0.26171875f, -0.03515625f}; // t=0.25
  const int  bh = (oh & 1) ? (oh >> 1) : (oh >> 1) - 1;
  const int  bw = (ow & 1) ? (ow >> 1) : (ow >> 1) - 1;
  const float* wh = (oh & 1) ? wo : we;
  const float* wv = (ow & 1) ? wo : we;

  float acc = 0.f;
#pragma unroll
  for (int i = 0; i < 4; ++i) {
    int rr = bh + i; rr = rr < 0 ? 0 : (rr > HH - 1 ? HH - 1 : rr);  // replicate
    float ra = 0.f;
#pragma unroll
    for (int j = 0; j < 4; ++j) {
      int cc = bw + j; cc = cc < 0 ? 0 : (cc > WW - 1 ? WW - 1 : cc);
      ra += wv[j] * src[rr * WW + cc];
    }
    acc += wh[i] * ra;
  }
  yu[idx] = acc;
}

// ------------------------- PAC gaussian kernel -----------------------------
// kern[b,t,h,w] = exp(-0.5 * sum_c (gcols - g)^2), normalized over 25 taps.
// g has a radius-2 zero halo -> all tap loads are immediate-offset, no checks.
__global__ void __launch_bounds__(256)
pac_kern_k(const float* __restrict__ g, float* __restrict__ kern) {
  int idx = blockIdx.x * blockDim.x + threadIdx.x;     // over B*HW2
  if (idx >= BB * HW2) return;
  const int b = idx / HW2, p = idx - b * HW2;
  const int hh = p / W2, ww = p - hh * W2;

  float s[25];
#pragma unroll
  for (int t = 0; t < 25; ++t) s[t] = 0.f;

  const float* gb = g + (size_t)b * CO * PHW2P + (size_t)(hh + 2) * PW2P + (ww + 2);
  for (int c = 0; c < CO; ++c) {
    const float* gc = gb + (size_t)c * PHW2P;
    const float ctr = gc[0];
#pragma unroll
    for (int t = 0; t < 25; ++t) {
      const float v = gc[(t / 5 - 2) * PW2P + (t % 5 - 2)];  // imm offsets
      const float d = v - ctr;
      s[t] += d * d;
    }
  }
  float e[25]; float tot = 0.f;
#pragma unroll
  for (int t = 0; t < 25; ++t) { e[t] = __expf(-0.5f * s[t]); tot += e[t]; }
  const float inv = 1.f / tot;
#pragma unroll
  for (int t = 0; t < 25; ++t)
    kern[((size_t)b * 25 + t) * HW2 + p] = e[t] * inv;
}

// ------------------------- launch ------------------------------------------
extern "C" void kernel_launch(void* const* d_in, const int* in_sizes, int n_in,
                              void* d_out, int out_size, void* d_ws, size_t ws_size,
                              hipStream_t stream) {
  (void)in_sizes; (void)n_in; (void)out_size; (void)ws_size;

  const float* input   = (const float*)d_in[0];
  const float* feature = (const float*)d_in[1];
  const float* conv1_w = (const float*)d_in[2];
  const float* conv1_b = (const float*)d_in[3];
  const float* in1_w   = (const float*)d_in[4];
  const float* in1_b   = (const float*)d_in[5];
  const float* prelu1  = (const float*)d_in[6];
  const float* conv2_w = (const float*)d_in[7];
  const float* conv2_b = (const float*)d_in[8];
  const float* in2_w   = (const float*)d_in[9];
  const float* in2_b   = (const float*)d_in[10];
  const float* prelu2  = (const float*)d_in[11];
  const float* res_w   = (const float*)d_in[12];
  const float* res_b   = (const float*)d_in[13];
  const float* feat_w  = (const float*)d_in[14];
  const float* feat_b  = (const float*)d_in[15];
  const float* pac_w   = (const float*)d_in[16];
  const float* pac_b   = (const float*)d_in[17];
  float* outp = (float*)d_out;

  // bump allocator over workspace (256B aligned slabs), ~42 MB total
  uint8_t* base = (uint8_t*)d_ws; size_t off = 0;
  auto alloc = [&](size_t bytes) -> void* {
    void* p = base + off; off += (bytes + 255) & ~(size_t)255; return p;
  };
  unsigned short* xb    = (unsigned short*)alloc((size_t)CIN * BB * PHW1 * 2);
  unsigned short* act1  = (unsigned short*)alloc((size_t)CO  * BB * PHW1 * 2);
  unsigned short* featb = (unsigned short*)alloc((size_t)CO  * N2 * 2);
  unsigned short* w1b   = (unsigned short*)alloc((size_t)CO * K1 * 2);
  unsigned short* w2b   = (unsigned short*)alloc((size_t)CO * K2 * 2);
  unsigned short* wrb   = (unsigned short*)alloc((size_t)CO * KR * 2);
  unsigned short* wfb   = (unsigned short*)alloc((size_t)CO * KF * 2);
  unsigned short* wpb   = (unsigned short*)alloc((size_t)CO * KP * 2);
  float* c1out  = (float*)alloc((size_t)CO * NL * 4);
  float* c2out  = (float*)alloc((size_t)CO * NL * 4);
  float* resout = (float*)alloc((size_t)CO * NL * 4);
  float* yf     = (float*)alloc((size_t)CO * NL * 4);
  float* yu     = (float*)alloc((size_t)BB * CO * PHW2P * 4);
  float* gbuf   = (float*)alloc((size_t)BB * CO * PHW2P * 4);  // padded guide
  float* kernb  = (float*)alloc((size_t)BB * 25 * HW2 * 4);

  const int T = 256;
  auto cdiv = [](int a, int b) { return (a + b - 1) / b; };

  // 1) pack activations (with zero halos) + weights (tap-major) to bf16
  permute_pad_k<<<cdiv(CIN * BB * PHW1, T), T, 0, stream>>>(input, xb, BB, CIN, HH, WW, 1);
  permute_pad_k<<<cdiv(CO * N2, T), T, 0, stream>>>(feature, featb, BB, CO, H2, W2, 0);
  zero_u16_k<<<cdiv(CO * BB * PHW1, T), T, 0, stream>>>(act1, CO * BB * PHW1);
  zero_f32_k<<<cdiv(BB * CO * PHW2P, T), T, 0, stream>>>(gbuf, BB * CO * PHW2P);
  repack_w_k<<<cdiv(CO * K1, T), T, 0, stream>>>(conv1_w, w1b, CO, CIN, 9);
  repack_w_k<<<cdiv(CO * K2, T), T, 0, stream>>>(conv2_w, w2b, CO, CO, 9);
  repack_w_k<<<cdiv(CO * KP, T), T, 0, stream>>>(pac_w,   wpb, CO, CO, 25);
  f32_to_bf16_k<<<cdiv(CO * KR, T), T, 0, stream>>>(res_w,  wrb, CO * KR);
  f32_to_bf16_k<<<cdiv(CO * KF, T), T, 0, stream>>>(feat_w, wfb, CO * KF);

  // 2) conv1 (implicit 3x3 GEMM, M=128 K=2304 N=4608) -> IN1 + PReLU -> bf16
  gemm_wmma_k<MODE_CONV3, NL, K1, CIN, HH, WW, 1, 0><<<NL / 64, T, 0, stream>>>(
      w1b, xb, nullptr, nullptr, conv1_b, c1out);
  instnorm_k<<<BB * CO, T, 0, stream>>>(c1out, nullptr, in1_w, in1_b, prelu1,
                                        nullptr, act1, BB, CO, HH, WW, 1);

  // 3) conv2 (K=1152) and residual 1x1 (K=256); IN2 + PReLU + residual -> y
  gemm_wmma_k<MODE_CONV3, NL, K2, CO, HH, WW, 1, 0><<<NL / 64, T, 0, stream>>>(
      w2b, act1, nullptr, nullptr, conv2_b, c2out);
  gemm_wmma_k<MODE_DIRECT, NL, KR, CIN, HH, WW, 1, 0><<<NL / 64, T, 0, stream>>>(
      wrb, xb, nullptr, nullptr, res_b, resout);
  instnorm_k<<<BB * CO, T, 0, stream>>>(c2out, resout, in2_w, in2_b, prelu2,
                                        yf, nullptr, BB, CO, HH, WW, 1);

  // 4) 2x bicubic upsample -> y_up (f32, radius-2 halo for PAC taps)
  upsample_bicubic_k<<<cdiv(BB * CO * PHW2P, T), T, 0, stream>>>(yf, yu);

  // 5) feat 1x1 GEMM (K=128, N=18432) -> g (radius-2 halo); gaussian kernel
  gemm_wmma_k<MODE_DIRECT, N2, KF, CO, H2, W2, 0, 2><<<N2 / 64, T, 0, stream>>>(
      wfb, featb, nullptr, nullptr, feat_b, gbuf);
  pac_kern_k<<<cdiv(BB * HW2, T), T, 0, stream>>>(gbuf, kernb);

  // 6) PAC GEMM (M=128, K=3200, N=18432) with kern folded into B on the fly
  gemm_wmma_k<MODE_PAC, N2, KP, CO, H2, W2, 2, 0><<<N2 / 64, T, 0, stream>>>(
      wpb, nullptr, yu, kernb, pac_b, outp);
}
